// MultiHeadAttention_635655160265
// MI455X (gfx1250) — matrix operand order repaired and checked by reading, not compiled, not run on hardware
//
#include <hip/hip_runtime.h>
#include <hip/hip_bf16.h>

// ---- problem constants (from reference) ----
#define BATCH   2
#define SEQ     2048
#define HID     1024
#define NHEAD   16
#define DHEAD   64
#define BS      (BATCH * SEQ)          // 4096 rows
#define OUT_ELEMS ((size_t)BS * HID)   // 4,194,304

typedef __attribute__((ext_vector_type(16))) _Float16 v16h;
typedef __attribute__((ext_vector_type(8)))  _Float16 v8h;
typedef __attribute__((ext_vector_type(8)))  float    v8f;
typedef __attribute__((ext_vector_type(4)))  float    v4f;
typedef __attribute__((ext_vector_type(4)))  unsigned int u32x4;
typedef __attribute__((ext_vector_type(8)))  int      i32x8;
typedef __attribute__((ext_vector_type(4)))  int      i32x4;

#define WMMA_F32_F16(a, b, acc) \
    __builtin_amdgcn_wmma_f32_16x16x32_f16(false, (a), false, (b), (short)0, (acc), false, false)

// ---------------------------------------------------------------------------
// WMMA fragment loaders per CDNA5 ISA VGPR layouts (CDNA5 ISA §7.12.2).
// All per-lane accesses are 16B contiguous -> global_load_b128 / ds_load_b128.
// ---------------------------------------------------------------------------

// A 16x32 f16 (row-major, leading dim lda):
// lanes 0-15: row M=lane, K={0..7,16..23}; lanes 16-31: row M=lane-16, K={8..15,24..31}
__device__ __forceinline__ v16h load_a_f16(const _Float16* __restrict__ A, int lda) {
    const int lane = threadIdx.x & 31;
    const _Float16* p = A + (size_t)(lane & 15) * lda + ((lane >> 4) << 3);
    const v8h lo = *(const v8h*)(p);
    const v8h hi = *(const v8h*)(p + 16);
    v16h a;
#pragma unroll
    for (int j = 0; j < 8; ++j) { a[j] = lo[j]; a[8 + j] = hi[j]; }
    return a;
}

// A fragment from f32 source (attn matrix), converted on load.
__device__ __forceinline__ v16h load_a_f32(const float* __restrict__ A, int lda) {
    const int lane = threadIdx.x & 31;
    const float* p = A + (size_t)(lane & 15) * lda + ((lane >> 4) << 3);
    const v4f a0 = *(const v4f*)(p);
    const v4f a1 = *(const v4f*)(p + 4);
    const v4f b0 = *(const v4f*)(p + 16);
    const v4f b1 = *(const v4f*)(p + 20);
    v16h a;
#pragma unroll
    for (int j = 0; j < 4; ++j) {
        a[j]      = (_Float16)a0[j];
        a[4 + j]  = (_Float16)a1[j];
        a[8 + j]  = (_Float16)b0[j];
        a[12 + j] = (_Float16)b1[j];
    }
    return a;
}

// B 32x16 f16 from a TRANSPOSED [N x K] source (leading dim ldm):
// lane = column N=lane&15, K={0..7,16..23} (lanes 0-15) / {8..15,24..31} (lanes 16-31) -> 2x b128 per lane.
__device__ __forceinline__ v16h load_bt_f16(const _Float16* __restrict__ Mp, int ldm) {
    const int lane = threadIdx.x & 31;
    const _Float16* p = Mp + (size_t)(lane & 15) * ldm + ((lane >> 4) << 3);
    const v8h lo = *(const v8h*)(p);
    const v8h hi = *(const v8h*)(p + 16);
    v16h b;
#pragma unroll
    for (int j = 0; j < 8; ++j) { b[j] = lo[j]; b[8 + j] = hi[j]; }
    return b;
}

// ---- LDS A-tile with TDM hardware padding: 8 DWORDs (16 halfs) inserted every 256 DWORDs
// (rows end up 2112 B apart -> 16-bank skew instead of a 16-way conflict).
// element e (halfs) -> padded offset e + 16 * (e / 512); 16B alignment preserved.
__device__ __forceinline__ int lds_pad(int e) { return e + ((e >> 9) << 4); }
#define ATILE_HALFS (16 * HID + (16 * HID / 512) * 16)   // 16896

__device__ __forceinline__ v16h load_a_lds(const _Float16* base, int row, int k0) {
    const int lane = threadIdx.x & 31;
    const int e = (lane & 15) * HID + k0 + ((lane >> 4) << 3);
    const v8h lo = *(const v8h*)(base + lds_pad(e));
    const v8h hi = *(const v8h*)(base + lds_pad(e + 16));
    v16h a;
#pragma unroll
    for (int j = 0; j < 8; ++j) { a[j] = lo[j]; a[8 + j] = hi[j]; }
    return a;
    (void)row;
}

// C/D 16x16 f32: lane holds column lane&15; VGPR r holds row r + 8*(lane>=16)
__device__ __forceinline__ void store_d_f32(float* __restrict__ C, int ldc, v8f c, float s) {
    const int lane  = threadIdx.x & 31;
    const int col   = lane & 15;
    const int rbase = (lane >> 4) << 3;
#pragma unroll
    for (int r = 0; r < 8; ++r) C[(size_t)(rbase + r) * ldc + col] = c[r] * s;
}

__device__ __forceinline__ void store_d_f32_nt(float* __restrict__ C, int ldc, v8f c, float s) {
    const int lane  = threadIdx.x & 31;
    const int col   = lane & 15;
    const int rbase = (lane >> 4) << 3;
#pragma unroll
    for (int r = 0; r < 8; ++r)
        __builtin_nontemporal_store(c[r] * s, &C[(size_t)(rbase + r) * ldc + col]);
}

__device__ __forceinline__ void store_d_f16(_Float16* __restrict__ C, int ldc, v8f c) {
    const int lane  = threadIdx.x & 31;
    const int col   = lane & 15;
    const int rbase = (lane >> 4) << 3;
#pragma unroll
    for (int r = 0; r < 8; ++r) C[(size_t)(rbase + r) * ldc + col] = (_Float16)c[r];
}

// Column-major f16 store: this lane's 8 rows are contiguous -> one b128 store.
__device__ __forceinline__ void store_d_f16_t(_Float16* __restrict__ Ccol, v8f c) {
    v8h o;
#pragma unroll
    for (int r = 0; r < 8; ++r) o[r] = (_Float16)c[r];
    *(v8h*)Ccol = o;
}

// ---------------------------------------------------------------------------
// TDM: DMA a contiguous 16-row x 1024-col f16 strip (32 KB) into LDS offset 0
// with hardware padding (pad_enable=1, interval=256 DW, amount=4 DW).
// Descriptor per CDNA5 ISA: count=1, type=2, data_size=2B,
// single row of 16384 elements (tile == tensor).
// ---------------------------------------------------------------------------
__device__ __forceinline__ void tdm_load_strip_to_lds(const _Float16* src) {
#if defined(__has_builtin) && __has_builtin(__builtin_amdgcn_tensor_load_to_lds)
    const unsigned long long ga = (unsigned long long)src;
    u32x4 g0;
    g0[0] = 1u;                                   // count=1, user mode
    g0[1] = 0u;                                   // lds_addr = base of LDS alloc
    g0[2] = (unsigned)(ga & 0xFFFFFFFFu);         // global_addr[31:0]
    g0[3] = (unsigned)((ga >> 32) & 0x1FFFFFFu) | (2u << 30); // addr[56:32] | type=2
    i32x8 g1;
    g1[0] = (1 << 16)                             // data_size = 2 bytes
          | (1 << 20)                             // pad_enable
          | (7 << 22)                             // pad_interval: 256 DWORDs
          | (3 << 25);                            // pad_amount: 4 DWORDs (16 B)
    g1[1] = (int)(16384u << 16);                  // tensor_dim0[15:0] << 16
    g1[2] = (int)(1u << 16);                      // tensor_dim0 hi=0 | tensor_dim1=1
    g1[3] = (int)(16384u << 16);                  // tensor_dim1 hi=0 | tile_dim0=16384
    g1[4] = 1;                                    // tile_dim1=1, tile_dim2=0
    g1[5] = 16384;                                // tensor_dim0_stride lo
    g1[6] = 0;
    g1[7] = 0;
    i32x4 z4 = {0, 0, 0, 0};
#if __has_include(<hip/amd_detail/amd_gfx1250_TDM.h>)
    i32x8 z8 = {0, 0, 0, 0, 0, 0, 0, 0};
    __builtin_amdgcn_tensor_load_to_lds(g0, g1, z4, z4, z8, 0); // therock 6-arg
#else
    __builtin_amdgcn_tensor_load_to_lds(g0, g1, z4, z4, 0);     // ROCm 7.2 5-arg
#endif
    __builtin_amdgcn_s_wait_tensorcnt(0);
#endif
}

// ---------------------------------------------------------------------------
// Elementwise f32 -> f16 (activations, row-major preserved)
// ---------------------------------------------------------------------------
__global__ __launch_bounds__(256) void cvt_f16(const float* __restrict__ src,
                                               _Float16* __restrict__ dst, int n) {
    int i = blockIdx.x * 256 + threadIdx.x;
    if (i < n) dst[i] = (_Float16)src[i];
}

// Transpose-convert weights: src [heads][K][N] f32 -> dst [heads*N][K] f16.
__global__ __launch_bounds__(256) void cvt_w_t(const float* __restrict__ src,
                                               _Float16* __restrict__ dst,
                                               int K, int N, int total) {
    int i = blockIdx.x * 256 + threadIdx.x;
    if (i >= total) return;
    const int hn = K * N;
    const int h = i / hn;
    const int r = i - h * hn;
    const int k = r / N;
    const int n = r - k * N;
    dst[((size_t)h * N + n) * K + k] = (_Float16)src[i];
}

// ---------------------------------------------------------------------------
// Shared GEMM: Y = X[4096][1024] @ W  (W given transposed: Wt [1024 out][1024 k])
// One block per 16-row strip (mt = blockIdx.x, 256 blocks). The A strip (32 KB)
// is DMA'd (padded) to LDS by the TDM once per block; 8 waves x 8 column tiles.
//   mode 0: f16 row-major out   mode 1: f32 row-major out
//   mode 2: f16 transposed out Vt[B][HID][SEQ] (V projection)
// ---------------------------------------------------------------------------
__global__ __launch_bounds__(256) void gemm_xw_kernel(const _Float16* __restrict__ X,
                                                      const _Float16* __restrict__ Wt,
                                                      void* __restrict__ Y, int mode) {
    __shared__ _Float16 Atile[ATILE_HALFS];       // 33 KB (padded)
    const int mt = blockIdx.x;
    const _Float16* strip = X + (size_t)mt * 16 * HID;

#if defined(__has_builtin) && __has_builtin(__builtin_amdgcn_tensor_load_to_lds)
    if (threadIdx.x < 32) tdm_load_strip_to_lds(strip);   // wave 0 drives the DMA
#else
    for (int j = threadIdx.x; j < 16 * HID / 8; j += 256)
        *(v8h*)(Atile + lds_pad(j * 8)) = ((const v8h*)strip)[j];
#endif
    __syncthreads();
    // The TDM writes Atile behind the compiler's back (descriptor holds only an
    // integer LDS offset). Escape the pointer + memory clobber so the ds_loads
    // below are not folded to undef / hoisted.
    {
        void* p = (void*)Atile;
        asm volatile("" : "+v"(p) :: "memory");
    }

    const int wv = threadIdx.x >> 5;              // 8 waves, 8 column tiles each
    v8f acc[8] = {};
    for (int k0 = 0; k0 < HID; k0 += 32) {
        const v16h a = load_a_lds(Atile, 0, k0);  // ds_load_b128 x2
        v16h bfrag[8];
#pragma unroll
        for (int i = 0; i < 8; ++i) {             // preload: 16 loads in flight
            const int nt = wv * 8 + i;
            __builtin_prefetch(Wt + (size_t)nt * 16 * HID + k0 + 32, 0, 3);
            bfrag[i] = load_bt_f16(Wt + (size_t)nt * 16 * HID + k0, HID);
        }
#pragma unroll
        for (int i = 0; i < 8; ++i)               // 8 independent WMMA chains
            acc[i] = WMMA_F32_F16(a, bfrag[i], acc[i]);
    }

#pragma unroll
    for (int i = 0; i < 8; ++i) {
        const int nt = wv * 8 + i;
        if (mode == 0) {
            store_d_f16((_Float16*)Y + (size_t)mt * 16 * HID + (size_t)nt * 16, HID, acc[i]);
        } else if (mode == 1) {
            store_d_f32((float*)Y + (size_t)mt * 16 * HID + (size_t)nt * 16, HID, acc[i], 1.0f);
        } else {
            const int bb   = (mt * 16) >> 11;     // batch of this row strip
            const int srow = (mt * 16) & 2047;
            const int lane = threadIdx.x & 31;
            _Float16* Ccol = (_Float16*)Y +
                ((size_t)bb * HID + (size_t)nt * 16 + (lane & 15)) * SEQ +
                srow + ((lane >> 4) << 3);
            store_d_f16_t(Ccol, acc[i]);          // one b128 store per lane
        }
    }
}

// ---------------------------------------------------------------------------
// scores[b,h,s,t] = (Q . K)/8, f32, nontemporal into d_out attn region.
// wave = ((b*16+h)*128 + st)*128 + tt ; Dh=64 -> 2 WMMAs per tile.
// ---------------------------------------------------------------------------
__global__ __launch_bounds__(256) void scores_kernel(const _Float16* __restrict__ Qh,
                                                     const _Float16* __restrict__ Kh,
                                                     float* __restrict__ attn) {
    const int wave = (blockIdx.x * 256 + threadIdx.x) >> 5;
    const int tt = wave & 127;
    const int st = (wave >> 7) & 127;
    const int bh = wave >> 14;
    const int h  = bh & 15;
    const int b  = bh >> 4;
    const _Float16* A = Qh + ((size_t)b * SEQ + (size_t)st * 16) * HID + h * DHEAD;
    const _Float16* M = Kh + ((size_t)b * SEQ + (size_t)tt * 16) * HID + h * DHEAD;
    const v16h a0 = load_a_f16(A, HID);
    const v16h b0 = load_bt_f16(M, HID);
    const v16h a1 = load_a_f16(A + 32, HID);
    const v16h b1 = load_bt_f16(M + 32, HID);
    v8f acc = {};
    acc = WMMA_F32_F16(a0, b0, acc);
    acc = WMMA_F32_F16(a1, b1, acc);
    float* C = attn + ((size_t)bh * SEQ + (size_t)st * 16) * SEQ + (size_t)tt * 16;
    store_d_f32_nt(C, SEQ, acc, 0.125f);           // streaming 537 MB: bypass caches
}

// ---------------------------------------------------------------------------
// In-place row softmax over 2048-wide rows; one 256-thread block per row.
// ---------------------------------------------------------------------------
__global__ __launch_bounds__(256) void softmax_kernel(float* __restrict__ attn) {
    __shared__ float red[256];
    float* p = attn + (size_t)blockIdx.x * SEQ;
    const int t = threadIdx.x;

    float m = -INFINITY;
    for (int i = t; i < SEQ; i += 256) m = fmaxf(m, p[i]);
    red[t] = m;
    __syncthreads();
    for (int s = 128; s > 0; s >>= 1) {
        if (t < s) red[t] = fmaxf(red[t], red[t + s]);
        __syncthreads();
    }
    m = red[0];
    __syncthreads();

    float sum = 0.f;
    for (int i = t; i < SEQ; i += 256) {
        float e = expf(p[i] - m);
        p[i] = e;
        sum += e;
    }
    red[t] = sum;
    __syncthreads();
    for (int s = 128; s > 0; s >>= 1) {
        if (t < s) red[t] += red[t + s];
        __syncthreads();
    }
    const float inv = 1.0f / red[0];
    for (int i = t; i < SEQ; i += 256)
        __builtin_nontemporal_store(p[i] * inv, &p[i]);
}

// ---------------------------------------------------------------------------
// wv = attn @ V, contracting t=2048 (64 WMMAs, 2 interleaved chains).
// V held transposed: Vt[B][HID][SEQ] -> two contiguous 16B K-runs per lane.
// wave = ((b*16+h)*128 + st)*4 + dt
// ---------------------------------------------------------------------------
__global__ __launch_bounds__(256) void attnv_kernel(const float* __restrict__ attn,
                                                    const _Float16* __restrict__ Vt,
                                                    _Float16* __restrict__ WV) {
    const int wave = (blockIdx.x * 256 + threadIdx.x) >> 5;
    const int dt = wave & 3;
    const int st = (wave >> 2) & 127;
    const int bh = wave >> 9;
    const int h  = bh & 15;
    const int b  = bh >> 4;
    const float*    A  = attn + ((size_t)bh * SEQ + (size_t)st * 16) * SEQ;
    const _Float16* Bb = Vt + ((size_t)b * HID + (size_t)h * DHEAD + dt * 16) * SEQ;
    v8f acc0 = {}, acc1 = {};
    for (int k0 = 0; k0 < SEQ; k0 += 64) {        // two K-steps per iteration
        __builtin_prefetch(A + k0 + 64, 0, 3);
        const v16h a0 = load_a_f32(A + k0, SEQ);
        const v16h b0 = load_bt_f16(Bb + k0, SEQ);
        const v16h a1 = load_a_f32(A + k0 + 32, SEQ);
        const v16h b1 = load_bt_f16(Bb + k0 + 32, SEQ);
        acc0 = WMMA_F32_F16(a0, b0, acc0);        // independent chains hide
        acc1 = WMMA_F32_F16(a1, b1, acc1);        // WMMA latency
    }
    v8f acc;
#pragma unroll
    for (int r = 0; r < 8; ++r) acc[r] = acc0[r] + acc1[r];
    store_d_f16(WV + ((size_t)b * SEQ + (size_t)st * 16) * HID + h * DHEAD + dt * 16,
                HID, acc);
}

// ---------------------------------------------------------------------------
// Workspace layout (_Float16 elements), 32M halfs = 64 MB total:
//  [0) WqT 1M [1M) WkT 1M [2M) WvT 1M [3M) WoT 1M
//  [4M) Xq 4M [8M) Xk 4M [12M) Xv 4M
//  [16M) Qh 4M [20M) Kh 4M [24M) Vt 4M [28M) WV 4M
// ---------------------------------------------------------------------------
extern "C" void kernel_launch(void* const* d_in, const int* in_sizes, int n_in,
                              void* d_out, int out_size, void* d_ws, size_t ws_size,
                              hipStream_t stream) {
    const float* query = (const float*)d_in[0];
    const float* key   = (const float*)d_in[1];
    const float* value = (const float*)d_in[2];
    const float* Wq    = (const float*)d_in[3];
    const float* Wk    = (const float*)d_in[4];
    const float* Wv    = (const float*)d_in[5];
    const float* Wo    = (const float*)d_in[6];

    float* out  = (float*)d_out;                 // [2,2048,1024]
    float* attn = (float*)d_out + OUT_ELEMS;     // [2,16,2048,2048]

    _Float16* w   = (_Float16*)d_ws;
    const size_t MW = (size_t)HID * HID;         // 1,048,576
    const size_t MX = (size_t)BS * HID;          // 4,194,304
    _Float16* wqt = w;
    _Float16* wkt = w + 1 * MW;
    _Float16* wvt = w + 2 * MW;
    _Float16* wot = w + 3 * MW;
    _Float16* xq  = w + 4 * MW;
    _Float16* xk  = xq + MX;
    _Float16* xv  = xk + MX;
    _Float16* qh  = xv + MX;
    _Float16* kh  = qh + MX;
    _Float16* vt  = kh + MX;
    _Float16* wvh = vt + MX;

    // weights: f32 -> f16 with transpose to [N][K]
    cvt_w_t<<<(int)(MW / 256), 256, 0, stream>>>(Wq, wqt, HID, DHEAD, (int)MW);
    cvt_w_t<<<(int)(MW / 256), 256, 0, stream>>>(Wk, wkt, HID, DHEAD, (int)MW);
    cvt_w_t<<<(int)(MW / 256), 256, 0, stream>>>(Wv, wvt, HID, DHEAD, (int)MW);
    cvt_w_t<<<(int)(MW / 256), 256, 0, stream>>>(Wo, wot, HID, HID,   (int)MW);
    // activations: f32 -> f16
    cvt_f16<<<(int)(MX / 256), 256, 0, stream>>>(query, xq, (int)MX);
    cvt_f16<<<(int)(MX / 256), 256, 0, stream>>>(key,   xk, (int)MX);
    cvt_f16<<<(int)(MX / 256), 256, 0, stream>>>(value, xv, (int)MX);

    // Q/K/V projections (TDM-staged A strips); V written transposed
    gemm_xw_kernel<<<256, 256, 0, stream>>>(xq, wqt, (void*)qh, 0);
    gemm_xw_kernel<<<256, 256, 0, stream>>>(xk, wkt, (void*)kh, 0);
    gemm_xw_kernel<<<256, 256, 0, stream>>>(xv, wvt, (void*)vt, 2);

    // QK^T / 8 : 524288 tiles -> 65536 blocks
    scores_kernel<<<65536, 256, 0, stream>>>(qh, kh, attn);

    // softmax over 65536 rows
    softmax_kernel<<<65536, 256, 0, stream>>>(attn);

    // attn @ V : 16384 tiles -> 2048 blocks
    attnv_kernel<<<2048, 256, 0, stream>>>(attn, vt, wvh);

    // output projection (f32 result)
    gemm_xw_kernel<<<256, 256, 0, stream>>>(wvh, wot, (void*)out, 1);
}
